// RegionProposalNetwork_66992899883387
// MI455X (gfx1250) — matrix operand order repaired and checked by reading, not compiled, not run on hardware
//
#include <hip/hip_runtime.h>

// ---------------- problem constants ----------------
#define NB        8
#define CIN       512
#define HH        75
#define WW        75
#define PIX       5625          // 75*75
#define NPIX_TOT  45000         // 8*5625
#define KTOT      4608          // 512*9
#define NANCH     50625         // 5625*9
#define NSORT     65536
#define N_PRE     12000
#define N_POST    2000
#define NEGV      (-1e30f)

// d_out layout (float elements)
#define OUT_LOCS    0            // 8*50625*4  = 1,620,000
#define OUT_SCORES  1620000      // 8*50625*2  =   810,000
#define OUT_ROIS    2430000      // 8*2000*4   =    64,000
#define OUT_IDX     2494000      // 8*2000     =    16,000
#define OUT_ANCHOR  2510000      // 50625*4    =   202,500

// workspace layout (bytes)
#define WS_H      ((size_t)0)           // bf16 h (pixel-major): 45000*512*2 = 46,080,000
#define WS_FG     ((size_t)46080000)    // float fg: 8*50625*4      =  1,620,000
#define WS_BOX    ((size_t)47700000)    // float boxes: 8*50625*4*4 =  6,480,000
#define WS_KEY    ((size_t)54180000)    // float keys: 8*65536*4    =  2,097,152
#define WS_IDXS   ((size_t)56277152)    // int   idxs: 8*65536*4    =  2,097,152
#define WS_WCOMB  ((size_t)58374304)    // bf16 Wcomb: 64*512*2     =     65,536
#define WS_W1     ((size_t)58439840)    // bf16 W1t: 512*4608*2     =  4,718,592
#define WS_XT     ((size_t)63158432)    // bf16 x_t (pixel-major): 45000*512*2 = 46,080,000

#define LSTR 80   // LDS row stride in ushorts (160B, 16B aligned rows)

typedef __attribute__((ext_vector_type(16))) __bf16 v16bf;
typedef __attribute__((ext_vector_type(8)))  float  v8f;

static __device__ inline unsigned short f2bf(float f) {
    unsigned int u = __float_as_uint(f);
    unsigned int r = (u + 0x7FFFu + ((u >> 16) & 1u)) >> 16;
    return (unsigned short)r;
}
static __device__ inline __bf16 bits2bf(unsigned short u) {
    union { unsigned short s; __bf16 b; } x; x.s = u; return x.b;
}
// generic LDS pointer -> 32-bit LDS offset (LDS aperture: low 32 bits)
static __device__ inline unsigned lds_off(const void* p) {
    return (unsigned)(size_t)p;
}
// async DMA: 16B global -> LDS, ASYNCcnt-tracked. IOFFSET adds to BOTH the
// LDS and global addresses (ISA 08_async_tensor §4.4), and our LDS/global
// segment strides match, so immediate offsets index both sides at once.
template <int OFF>
static __device__ inline void async_b128(unsigned lds, const void* gptr) {
    unsigned long long ga = (unsigned long long)gptr;
    asm volatile("global_load_async_to_lds_b128 %0, %1, off offset:%2"
                 :: "v"(lds), "v"(ga), "i"(OFF) : "memory");
}
static __device__ inline void wait_async0() {
    asm volatile("s_wait_asynccnt 0x0" ::: "memory");
}
static __device__ inline void zero16(unsigned short* d) {
    *(uint4*)d = make_uint4(0u, 0u, 0u, 0u);
}

// ---------------- prep kernels ----------------
// W1 (512,512,3,3) fp32 -> Aw[m][k] bf16, k = (ky*3+kx)*512 + c
__global__ void prep_w1_k(const float* __restrict__ W1, unsigned short* __restrict__ Aw) {
    int g = blockIdx.x * blockDim.x + threadIdx.x;
    if (g >= CIN * KTOT) return;
    int m = g / KTOT, k = g % KTOT;
    int kpos = k >> 9, c = k & 511;
    int ky = kpos / 3, kx = kpos % 3;
    Aw[(size_t)m * KTOT + k] = f2bf(W1[(((size_t)m * CIN + c) * 3 + ky) * 3 + kx]);
}
// Ws(18,512), Wl(36,512) -> Wcomb[64][512] bf16 (rows 54..63 zero)
__global__ void prep_wcomb_k(const float* __restrict__ Ws, const float* __restrict__ Wl,
                             unsigned short* __restrict__ Wc) {
    int g = blockIdx.x * blockDim.x + threadIdx.x;
    if (g >= 64 * CIN) return;
    int m = g / CIN, c = g % CIN;
    float v = 0.f;
    if (m < 18)      v = Ws[(size_t)m * CIN + c];
    else if (m < 54) v = Wl[(size_t)(m - 18) * CIN + c];
    Wc[g] = f2bf(v);
}
// x NCHW fp32 -> x_t[(n*PIX+pi)*CIN + c] bf16 (pixel-major, contiguous channels)
__global__ void prep_x_k(const float* __restrict__ x, unsigned short* __restrict__ xt) {
    int g = blockIdx.x * blockDim.x + threadIdx.x;
    if (g >= NPIX_TOT * CIN) return;
    int pg = g / CIN, c = g % CIN;
    int n = pg / PIX, pi = pg % PIX;
    xt[g] = f2bf(x[((size_t)n * CIN + c) * PIX + pi]);
}

// ---------------- conv3x3 + bias + relu via WMMA (implicit GEMM) ----------------
// grid (1407, 4), block 256 (8 waves). Tile: 128 M x 32 N.
// Loop nest: 9 taps x 8 channel-chunks of 64 (4 WMMA per chunk).
__global__ __launch_bounds__(256) void conv3x3_wmma_k(
    const unsigned short* __restrict__ xt, const unsigned short* __restrict__ Aw,
    const float* __restrict__ b1, unsigned short* __restrict__ hout)
{
    const int tid  = threadIdx.x;
    const int lane = tid & 31;
    const int wave = tid >> 5;
    const int p0   = blockIdx.x * 32;
    const int m0   = blockIdx.y * 128;

    __shared__ __align__(16) unsigned short As[128 * LSTR];  // 20,480 B
    __shared__ __align__(16) unsigned short Bs[32 * LSTR];   //  5,120 B

    v8f acc0 = {}; v8f acc1 = {};

    // A-staging: 2 threads/row, 4 x 16B segments each (row = 64 bf16 = 128 B)
    const int arow  = tid >> 1;
    const int aseg0 = (tid & 1) * 4;
    // B-staging: 8 threads/pixel, 1 x 16B segment each
    const int bpix = tid >> 3;
    const int bseg = tid & 7;
    const int pg   = p0 + bpix;
    const bool pvalid = pg < NPIX_TOT;
    int bn = 0, by = 0, bx = 0;
    if (pvalid) { bn = pg / PIX; int pi = pg % PIX; by = pi / WW; bx = pi % WW; }

    // invariant base addresses (LDS fixed, global bumped by pointer math)
    const unsigned short* aG = Aw + (size_t)(m0 + arow) * KTOT + aseg0 * 8;
    const unsigned aL = lds_off(&As[arow * LSTR + aseg0 * 8]);
    const unsigned bL = lds_off(&Bs[bpix * LSTR + bseg * 8]);
    unsigned short* bZ = &Bs[bpix * LSTR + bseg * 8];

    const int rowm = lane & 15;
    const int kg   = lane >> 4;
    const unsigned short* aFrag = &As[(wave * 16 + rowm) * LSTR];
    const unsigned short* bFrag0 = &Bs[rowm * LSTR];
    const unsigned short* bFrag1 = &Bs[(16 + rowm) * LSTR];

    for (int tap = 0; tap < 9; ++tap) {
        const int ky = tap / 3, kx = tap - (tap / 3) * 3;
        const int iy = by + ky - 1;
        const int ix = bx + kx - 1;
        const bool inb = pvalid && iy >= 0 && iy < HH && ix >= 0 && ix < WW;
        const unsigned short* bG =
            xt + ((size_t)bn * PIX + iy * WW + ix) * CIN + bseg * 8;
        if (!inb) zero16(bZ);                 // once per tap; nobody else writes it
        __builtin_prefetch(aG + CIN, 0, 1);   // speculative: next tap's A rows

        for (int cc = 0; cc < CIN; cc += 64) {
            // stage A tile 128x64: 4 segments, shared base, IOFFSET-indexed
            async_b128<0>(aL, aG);
            async_b128<16>(aL, aG);
            async_b128<32>(aL, aG);
            async_b128<48>(aL, aG);
            // stage B tile 32x64 (1 segment/thread)
            if (inb) async_b128<0>(bL, bG);
            wait_async0();
            __syncthreads();

            #pragma unroll
            for (int ch = 0; ch < 2; ++ch) {
                const int kc = ch * 32;
                v16bf af, bf0, bf1;
                #pragma unroll
                for (int e = 0; e < 16; ++e) {
                    const int vi = e >> 1, sub = e & 1;
                    const int ka = kc + kg * 16 + e;
                    const int kb = kc + kg * 16 + e;
                    af[e]  = bits2bf(aFrag[ka]);
                    bf0[e] = bits2bf(bFrag0[kb]);
                    bf1[e] = bits2bf(bFrag1[kb]);
                }
                acc0 = __builtin_amdgcn_wmma_f32_16x16x32_bf16(false, af, false, bf0,
                                                               (short)0, acc0, false, false);
                acc1 = __builtin_amdgcn_wmma_f32_16x16x32_bf16(false, af, false, bf1,
                                                               (short)0, acc1, false, false);
            }
            __syncthreads();
            aG += 64; bG += 64;
        }
    }

    // epilogue: bias + relu, store bf16 h pixel-major: h[pg*512 + m]
    const int colp = lane & 15;
    #pragma unroll
    for (int r = 0; r < 8; ++r) {
        const int m = m0 + wave * 16 + r + kg * 8;
        const float bias = b1[m];
        int pga = p0 + colp;
        if (pga < NPIX_TOT) {
            float v = acc0[r] + bias; v = v > 0.f ? v : 0.f;
            hout[(size_t)pga * CIN + m] = f2bf(v);
        }
        int pgb = p0 + 16 + colp;
        if (pgb < NPIX_TOT) {
            float v = acc1[r] + bias; v = v > 0.f ? v : 0.f;
            hout[(size_t)pgb * CIN + m] = f2bf(v);
        }
    }
}

// ---------------- fused 1x1 convs (scores+locs) via WMMA ----------------
// grid (1407), block 128 (4 waves). Tile: 64 M x 32 N, K=512, chunk 64.
__global__ __launch_bounds__(128) void head1x1_wmma_k(
    const unsigned short* __restrict__ h, const unsigned short* __restrict__ Wc,
    const float* __restrict__ bs, const float* __restrict__ bl,
    float* __restrict__ out_locs, float* __restrict__ out_scores)
{
    const int tid  = threadIdx.x;
    const int lane = tid & 31;
    const int wave = tid >> 5;
    const int p0   = blockIdx.x * 32;

    __shared__ __align__(16) unsigned short As[64 * LSTR];   // 10,240 B
    __shared__ __align__(16) unsigned short Bs[32 * LSTR];   //  5,120 B

    v8f acc0 = {}; v8f acc1 = {};

    const int arow  = tid >> 1;         // 0..63
    const int aseg0 = (tid & 1) * 4;
    const int bpix  = tid >> 2;         // 0..31
    const int bseg0 = (tid & 3) * 2;
    const int pg    = p0 + bpix;
    const bool pvalid = pg < NPIX_TOT;

    const unsigned short* aG = Wc + (size_t)arow * CIN + aseg0 * 8;
    const unsigned aL = lds_off(&As[arow * LSTR + aseg0 * 8]);
    const unsigned short* bG = h + (size_t)pg * CIN + bseg0 * 8;
    const unsigned bL = lds_off(&Bs[bpix * LSTR + bseg0 * 8]);

    if (!pvalid) { zero16(&Bs[bpix * LSTR + bseg0 * 8]); zero16(&Bs[bpix * LSTR + bseg0 * 8 + 8]); }

    const int rowm = lane & 15;
    const int kg   = lane >> 4;
    const unsigned short* aFrag = &As[(wave * 16 + rowm) * LSTR];
    const unsigned short* bFrag0 = &Bs[rowm * LSTR];
    const unsigned short* bFrag1 = &Bs[(16 + rowm) * LSTR];

    for (int kk = 0; kk < CIN; kk += 64) {
        async_b128<0>(aL, aG);
        async_b128<16>(aL, aG);
        async_b128<32>(aL, aG);
        async_b128<48>(aL, aG);
        if (pvalid) {
            async_b128<0>(bL, bG);
            async_b128<16>(bL, bG);
        }
        wait_async0();
        __syncthreads();

        #pragma unroll
        for (int ch = 0; ch < 2; ++ch) {
            const int kc = ch * 32;
            v16bf af, bf0, bf1;
            #pragma unroll
            for (int e = 0; e < 16; ++e) {
                const int vi = e >> 1, sub = e & 1;
                const int ka = kc + kg * 16 + e;
                const int kb = kc + kg * 16 + e;
                af[e]  = bits2bf(aFrag[ka]);
                bf0[e] = bits2bf(bFrag0[kb]);
                bf1[e] = bits2bf(bFrag1[kb]);
            }
            acc0 = __builtin_amdgcn_wmma_f32_16x16x32_bf16(false, af, false, bf0,
                                                           (short)0, acc0, false, false);
            acc1 = __builtin_amdgcn_wmma_f32_16x16x32_bf16(false, af, false, bf1,
                                                           (short)0, acc1, false, false);
        }
        __syncthreads();
        aG += 64; bG += 64;
    }

    const int colp = lane & 15;
    #pragma unroll
    for (int r = 0; r < 8; ++r) {
        const int m = wave * 16 + r + kg * 8;   // 0..63
        #pragma unroll
        for (int st = 0; st < 2; ++st) {
            const int pgc = p0 + st * 16 + colp;
            if (pgc >= NPIX_TOT) continue;
            const float a = (st == 0) ? acc0[r] : acc1[r];
            const int n = pgc / PIX, pi = pgc % PIX;
            if (m < 18) {
                out_scores[(size_t)n * (NANCH * 2) + (size_t)pi * 18 + m] = a + bs[m];
            } else if (m < 54) {
                const int j = m - 18;
                out_locs[(size_t)n * (NANCH * 4) + (size_t)pi * 36 + j] = a + bl[j];
            }
        }
    }
}

// ---------------- softmax fg score ----------------
__global__ void softmax_fg_k(const float* __restrict__ scores, float* __restrict__ fg) {
    int g = blockIdx.x * blockDim.x + threadIdx.x;
    if (g >= NB * NANCH) return;
    int n = g / NANCH, t = g % NANCH;
    const float* s = scores + (size_t)n * (NANCH * 2) + (size_t)t * 2;
    float s0 = s[0], s1 = s[1];
    float m = s0 > s1 ? s0 : s1;
    float e0 = __expf(s0 - m), e1 = __expf(s1 - m);
    fg[(size_t)n * NANCH + t] = e1 / (e0 + e1);
}

static __device__ inline void anchor_at(int g, float* a4) {
    const int p = g / 9, aa = g % 9;
    const int y = p / WW, xx = p % WW;
    const float ratios[3] = {0.5f, 1.f, 2.f};
    const float scales[3] = {8.f, 16.f, 32.f};
    const float r = ratios[aa / 3], s = scales[aa % 3];
    const float hh = 16.f * s * __fsqrt_rn(r);
    const float ww = 16.f * s * __fsqrt_rn(1.f / r);
    const float sy = (float)y * 16.f, sx = (float)xx * 16.f;
    a4[0] = 8.f - hh * 0.5f + sy;
    a4[1] = 8.f - ww * 0.5f + sx;
    a4[2] = 8.f + hh * 0.5f + sy;
    a4[3] = 8.f + ww * 0.5f + sx;
}

__global__ void anchors_k(float* __restrict__ out_anchor) {
    int g = blockIdx.x * blockDim.x + threadIdx.x;
    if (g >= NANCH) return;
    float a4[4]; anchor_at(g, a4);
    out_anchor[(size_t)g * 4 + 0] = a4[0];
    out_anchor[(size_t)g * 4 + 1] = a4[1];
    out_anchor[(size_t)g * 4 + 2] = a4[2];
    out_anchor[(size_t)g * 4 + 3] = a4[3];
}

// ---------------- proposal decode + clip + min-size filter ----------------
__global__ void proposal_k(const float* __restrict__ locs, const float* __restrict__ fg,
                           float* __restrict__ keys, int* __restrict__ idxs,
                           float* __restrict__ boxes,
                           const int* __restrict__ img_h, const int* __restrict__ img_w) {
    int g = blockIdx.x * blockDim.x + threadIdx.x;
    if (g >= NB * NSORT) return;
    const int n = g >> 16, t = g & (NSORT - 1);
    if (t >= NANCH) { keys[(size_t)n * NSORT + t] = -3.0e38f; idxs[(size_t)n * NSORT + t] = 0; return; }

    float a4[4]; anchor_at(t, a4);
    const float* l = locs + (size_t)n * (NANCH * 4) + (size_t)t * 4;
    const float ah = a4[2] - a4[0], aw = a4[3] - a4[1];
    const float cy = a4[0] + 0.5f * ah, cx = a4[1] + 0.5f * aw;
    const float ncy = l[0] * ah + cy, ncx = l[1] * aw + cx;
    const float nh = __expf(l[2]) * ah, nw = __expf(l[3]) * aw;
    const float fh = (float)img_h[0], fw = (float)img_w[0];
    float y1 = ncy - 0.5f * nh, x1 = ncx - 0.5f * nw;
    float y2 = ncy + 0.5f * nh, x2 = ncx + 0.5f * nw;
    y1 = fminf(fmaxf(y1, 0.f), fh); x1 = fminf(fmaxf(x1, 0.f), fw);
    y2 = fminf(fmaxf(y2, 0.f), fh); x2 = fminf(fmaxf(x2, 0.f), fw);

    float sc = fg[(size_t)n * NANCH + t];
    const float hs = y2 - y1, ws2 = x2 - x1;
    if (!(hs >= 16.f && ws2 >= 16.f)) sc = NEGV;

    keys[(size_t)n * NSORT + t] = sc;
    idxs[(size_t)n * NSORT + t] = t;
    float* b = boxes + ((size_t)n * NANCH + t) * 4;
    b[0] = y1; b[1] = x1; b[2] = y2; b[3] = x2;
}

// ---------------- bitonic sort (descending) ----------------
__global__ void bitonic_pass_k(float* __restrict__ keys, int* __restrict__ idxs, int j, int k) {
    const int i = blockIdx.x * blockDim.x + threadIdx.x;
    const size_t base = (size_t)blockIdx.y * NSORT;
    const int ixj = i ^ j;
    if (ixj <= i) return;
    const bool blkUp = ((i & k) == 0);
    float ki = keys[base + i], kj = keys[base + ixj];
    const bool doswap = blkUp ? (ki < kj) : (ki > kj);
    if (doswap) {
        keys[base + i] = kj; keys[base + ixj] = ki;
        int ti = idxs[base + i];
        idxs[base + i] = idxs[base + ixj];
        idxs[base + ixj] = ti;
    }
}

// ---------------- greedy NMS over sorted top-12000 ----------------
__global__ __launch_bounds__(256) void nms_k(const float* __restrict__ keys,
                                             const int* __restrict__ idxs,
                                             const float* __restrict__ boxes,
                                             float* __restrict__ out_rois,
                                             float* __restrict__ out_idx) {
    const int n   = blockIdx.x;
    const int tid = threadIdx.x;
    const size_t base = (size_t)n * NSORT;

    __shared__ unsigned char sup[N_PRE];
    __shared__ float cb[4];
    __shared__ int   sptr, svalid, scur;

    for (int i = tid; i < N_PRE; i += blockDim.x) sup[i] = 0;
    if (tid == 0) sptr = 0;
    __syncthreads();

    for (int it = 0; it < N_POST; ++it) {
        if (tid == 0) {
            int p = sptr;
            while (p < N_PRE && sup[p]) ++p;
            int v = 0;
            if (p < N_PRE && keys[base + p] > -1e29f) v = 1;
            svalid = v; scur = p;
            if (v) {
                const int orig = idxs[base + p];
                const float* b = boxes + ((size_t)n * NANCH + orig) * 4;
                cb[0] = b[0]; cb[1] = b[1]; cb[2] = b[2]; cb[3] = b[3];
            } else {
                cb[0] = 0.f; cb[1] = 0.f; cb[2] = 0.f; cb[3] = 0.f;
            }
            sptr = p + 1;
        }
        __syncthreads();
        if (tid < 4)  out_rois[((size_t)n * N_POST + it) * 4 + tid] = cb[tid];
        if (tid == 4) out_idx[(size_t)n * N_POST + it] = (float)n;
        if (svalid) {
            const float y1 = cb[0], x1 = cb[1], y2 = cb[2], x2 = cb[3];
            const float areaK = (y2 - y1) * (x2 - x1);
            for (int c = scur + 1 + tid; c < N_PRE; c += blockDim.x) {
                if (sup[c]) continue;
                const int orig = idxs[base + c];
                const float* b = boxes + ((size_t)n * NANCH + orig) * 4;
                const float iy1 = fmaxf(b[0], y1), ix1 = fmaxf(b[1], x1);
                const float iy2 = fminf(b[2], y2), ix2 = fminf(b[3], x2);
                const float inter = fmaxf(iy2 - iy1, 0.f) * fmaxf(ix2 - ix1, 0.f);
                const float areaC = (b[2] - b[0]) * (b[3] - b[1]);
                const float iou = inter / (areaC + areaK - inter + 1e-9f);
                if (iou > 0.7f) sup[c] = 1;
            }
        }
        __syncthreads();
    }
}

// ---------------- launcher ----------------
extern "C" void kernel_launch(void* const* d_in, const int* in_sizes, int n_in,
                              void* d_out, int out_size, void* d_ws, size_t ws_size,
                              hipStream_t stream) {
    (void)in_sizes; (void)n_in; (void)out_size; (void)ws_size;
    const float* x   = (const float*)d_in[0];
    const float* W1  = (const float*)d_in[1];
    const float* b1  = (const float*)d_in[2];
    const float* Ws  = (const float*)d_in[3];
    const float* bs  = (const float*)d_in[4];
    const float* Wl  = (const float*)d_in[5];
    const float* bl  = (const float*)d_in[6];
    const int* img_h = (const int*)d_in[7];
    const int* img_w = (const int*)d_in[8];

    float* out = (float*)d_out;
    char*  ws  = (char*)d_ws;

    unsigned short* h     = (unsigned short*)(ws + WS_H);
    float*          fg    = (float*)(ws + WS_FG);
    float*          boxes = (float*)(ws + WS_BOX);
    float*          keys  = (float*)(ws + WS_KEY);
    int*            idxs  = (int*)(ws + WS_IDXS);
    unsigned short* Wc    = (unsigned short*)(ws + WS_WCOMB);
    unsigned short* Aw    = (unsigned short*)(ws + WS_W1);
    unsigned short* xt    = (unsigned short*)(ws + WS_XT);

    prep_w1_k<<<(CIN * KTOT + 255) / 256, 256, 0, stream>>>(W1, Aw);
    prep_wcomb_k<<<(64 * CIN + 255) / 256, 256, 0, stream>>>(Ws, Wl, Wc);
    prep_x_k<<<(NPIX_TOT * CIN + 255) / 256, 256, 0, stream>>>(x, xt);

    const int ntile = (NPIX_TOT + 31) / 32;  // 1407
    conv3x3_wmma_k<<<dim3(ntile, 4), 256, 0, stream>>>(xt, Aw, b1, h);
    head1x1_wmma_k<<<dim3(ntile), 128, 0, stream>>>(h, Wc, bs, bl,
                                                    out + OUT_LOCS, out + OUT_SCORES);

    softmax_fg_k<<<(NB * NANCH + 255) / 256, 256, 0, stream>>>(out + OUT_SCORES, fg);
    anchors_k<<<(NANCH + 255) / 256, 256, 0, stream>>>(out + OUT_ANCHOR);
    proposal_k<<<(NB * NSORT) / 256, 256, 0, stream>>>(out + OUT_LOCS, fg, keys, idxs,
                                                       boxes, img_h, img_w);

    for (int k = 2; k <= NSORT; k <<= 1)
        for (int j = k >> 1; j > 0; j >>= 1)
            bitonic_pass_k<<<dim3(NSORT / 256, NB), 256, 0, stream>>>(keys, idxs, j, k);

    nms_k<<<NB, 256, 0, stream>>>(keys, idxs, boxes, out + OUT_ROIS, out + OUT_IDX);
}
